// GCN_3layer_61830349193499
// MI455X (gfx1250) — compile-verified, matrix operand order checked by reading, not run on hardware
//
#include <hip/hip_runtime.h>
#include <hip/hip_bf16.h>

typedef __attribute__((ext_vector_type(16))) __bf16 v16bf;
typedef __attribute__((ext_vector_type(8)))  float  v8f;

// ---------------------------------------------------------------------------
// Degree / normalization kernels (PyG GCN symmetric norm with self-loops)
// ---------------------------------------------------------------------------
__global__ void deg_init_kernel(float* __restrict__ deg, int n) {
    int i = blockIdx.x * blockDim.x + threadIdx.x;
    if (i < n) deg[i] = 1.0f;                      // self-loop contributes 1
}

__global__ void deg_count_kernel(const int* __restrict__ dst,
                                 float* __restrict__ deg, int E) {
    int i = blockIdx.x * blockDim.x + threadIdx.x;
    if (i < E) unsafeAtomicAdd(&deg[dst[i]], 1.0f);
}

__global__ void deg_rsqrt_kernel(const float* __restrict__ deg,
                                 float* __restrict__ isq, int n) {
    int i = blockIdx.x * blockDim.x + threadIdx.x;
    if (i < n) isq[i] = rsqrtf(deg[i]);            // deg >= 1 always
}

// ---------------------------------------------------------------------------
// WMMA GEMM: H[n x COUT] = X[n x CIN] @ W[CIN x COUT], fp32 in/out,
// bf16 multiply + fp32 accumulate on v_wmma_f32_16x16x32_bf16.
// One wave (32 lanes) computes one 16x16 output tile. blockDim=(32, COUT/16).
// CIN/COUT are compile-time: K-loop fully unrolls -> 4 back-to-back WMMAs
// with all global loads hoisted ahead of the MMA chain.
// Requires n % 16 == 0 (N=10000 -> 625 row tiles) so EXEC is all-ones.
// ---------------------------------------------------------------------------
template <int CIN, int COUT>
__global__ void gemm_wmma_bf16_kernel(const float* __restrict__ X,
                                      const float* __restrict__ W,
                                      float* __restrict__ H) {
    const int lane = threadIdx.x;            // 0..31
    const int ctile = threadIdx.y;           // which 16-col tile
    const int m0 = blockIdx.x * 16;
    const int half = lane >> 4;              // 0: K {0..7,16..23}; 1: K {8..15,24..31}
    const int l = lane & 15;
    const int m = m0 + l;                    // A-matrix row for this lane
    const int n = ctile * 16 + l;            // B-matrix col for this lane

    v8f acc = {0.f, 0.f, 0.f, 0.f, 0.f, 0.f, 0.f, 0.f};

    #pragma unroll
    for (int kb = 0; kb < CIN; kb += 32) {
        // A tile row: contiguous fp32, two float4 loads per K-half
        float xa[16];
        const float4* xr0 = reinterpret_cast<const float4*>(&X[(size_t)m * CIN + kb + half * 8]);
        const float4* xr1 = reinterpret_cast<const float4*>(&X[(size_t)m * CIN + kb + 16 + half * 8]);
        *reinterpret_cast<float4*>(&xa[0])  = xr0[0];
        *reinterpret_cast<float4*>(&xa[4])  = xr0[1];
        *reinterpret_cast<float4*>(&xa[8])  = xr1[0];
        *reinterpret_cast<float4*>(&xa[12]) = xr1[1];

        v16bf a, b;
        #pragma unroll
        for (int e = 0; e < 8; ++e) {
            const int k0 = kb + half * 8 + e;        // K for elements 0..7
            const int k1 = k0 + 16;                  // K for elements 8..15
            a[e]     = (__bf16)xa[e];
            a[e + 8] = (__bf16)xa[e + 8];
            b[e]     = (__bf16)W[(size_t)k0 * COUT + n];
            b[e + 8] = (__bf16)W[(size_t)k1 * COUT + n];
        }
        acc = __builtin_amdgcn_wmma_f32_16x16x32_bf16(
            /*neg_a=*/false, a, /*neg_b=*/false, b,
            /*c_mod=*/(short)0, acc, /*reuse_a=*/false, /*reuse_b=*/false);
    }

    // D layout: VGPR v holds (M = v + 8*half, N = lane&15)
    #pragma unroll
    for (int v = 0; v < 8; ++v) {
        const int r = m0 + v + half * 8;
        H[(size_t)r * COUT + ctile * 16 + l] = acc[v];
    }
}

// ---------------------------------------------------------------------------
// agg[i,c] = h[i,c] * isq[i]^2  (self-loop term; also initializes agg buffer)
// cshift = log2(C)
// ---------------------------------------------------------------------------
__global__ void self_init_kernel(const float* __restrict__ H,
                                 const float* __restrict__ isq,
                                 float* __restrict__ AGG,
                                 long total, int cshift) {
    long i = (long)blockIdx.x * blockDim.x + threadIdx.x;
    if (i < total) {
        int node = (int)(i >> cshift);
        float s = isq[node];
        AGG[i] = H[i] * s * s;
    }
}

// ---------------------------------------------------------------------------
// Edge scatter: agg[dst] += h[src] * isq[src]*isq[dst]
// 4 channels per thread (float4 gather + 4 HW fp32 atomics).
// lshift = log2(C/4) = lanes-per-edge shift.
// ---------------------------------------------------------------------------
__global__ void scatter_edges_kernel(const float* __restrict__ H,
                                     const int* __restrict__ src,
                                     const int* __restrict__ dst,
                                     const float* __restrict__ isq,
                                     float* __restrict__ AGG,
                                     long total, int C, int lshift) {
    long idx = (long)blockIdx.x * blockDim.x + threadIdx.x;
    if (idx >= total) return;
    const int e = (int)(idx >> lshift);
    const int c = (int)(idx & ((1 << lshift) - 1)) << 2;
    const int s = src[e];
    const int d = dst[e];
    const float nrm = isq[s] * isq[d];
    const float4 v = *reinterpret_cast<const float4*>(&H[(size_t)s * C + c]);
    float* out = &AGG[(size_t)d * C + c];
    unsafeAtomicAdd(out + 0, v.x * nrm);
    unsafeAtomicAdd(out + 1, v.y * nrm);
    unsafeAtomicAdd(out + 2, v.z * nrm);
    unsafeAtomicAdd(out + 3, v.w * nrm);
}

// ---------------------------------------------------------------------------
// out[i,c] = act(agg[i,c] + b[c]); cmask = C-1 (C power of two)
// ---------------------------------------------------------------------------
__global__ void bias_act_kernel(const float* __restrict__ AGG,
                                const float* __restrict__ bias,
                                float* __restrict__ OUT,
                                long total, int cmask, int do_relu) {
    long i = (long)blockIdx.x * blockDim.x + threadIdx.x;
    if (i < total) {
        float v = AGG[i] + bias[(int)(i & cmask)];
        OUT[i] = do_relu ? fmaxf(v, 0.0f) : v;
    }
}

// ---------------------------------------------------------------------------
// Launch: 3-layer GCN
// inputs: x[N,128], edge_index[2,E] (int32), W1[128,128], b1, W2[128,128], b2,
//         W3[128,64], b3. output: [N,64] fp32.
// ---------------------------------------------------------------------------
extern "C" void kernel_launch(void* const* d_in, const int* in_sizes, int n_in,
                              void* d_out, int out_size, void* d_ws, size_t ws_size,
                              hipStream_t stream) {
    const float* x  = (const float*)d_in[0];
    const int*   ei = (const int*)d_in[1];
    const float* W1 = (const float*)d_in[2];
    const float* b1 = (const float*)d_in[3];
    const float* W2 = (const float*)d_in[4];
    const float* b2 = (const float*)d_in[5];
    const float* W3 = (const float*)d_in[6];
    const float* b3 = (const float*)d_in[7];

    const int CIN = 128, CH = 128, COUT = 64;
    const int n = in_sizes[0] / CIN;        // 10000
    const int E = in_sizes[1] / 2;          // 640000
    const int* src = ei;
    const int* dst = ei + E;

    float* ws   = (float*)d_ws;
    float* deg  = ws;
    float* isq  = deg  + n;
    float* hlin = isq  + n;
    float* agg  = hlin + (size_t)n * CH;
    float* feat = agg  + (size_t)n * CH;

    const int TB = 256;
    // --- normalization ---
    deg_init_kernel <<<(n + TB - 1) / TB, TB, 0, stream>>>(deg, n);
    deg_count_kernel<<<(E + TB - 1) / TB, TB, 0, stream>>>(dst, deg, E);
    deg_rsqrt_kernel<<<(n + TB - 1) / TB, TB, 0, stream>>>(deg, isq, n);

    const int row_tiles = n / 16;           // 625

    // --- layer 1: x -> feat (relu) ---
    {
        const int C = CH;                   // 128
        gemm_wmma_bf16_kernel<128, 128><<<row_tiles, dim3(32, C / 16), 0, stream>>>(x, W1, hlin);
        long nc = (long)n * C;
        self_init_kernel<<<(int)((nc + TB - 1) / TB), TB, 0, stream>>>(hlin, isq, agg, nc, 7);
        long tot = (long)E << 5;            // C/4 = 32 lanes per edge
        scatter_edges_kernel<<<(int)((tot + TB - 1) / TB), TB, 0, stream>>>(
            hlin, src, dst, isq, agg, tot, C, 5);
        bias_act_kernel<<<(int)((nc + TB - 1) / TB), TB, 0, stream>>>(agg, b1, feat, nc, C - 1, 1);
    }

    // --- layer 2: feat -> feat (relu) ---
    {
        const int C = CH;                   // 128
        gemm_wmma_bf16_kernel<128, 128><<<row_tiles, dim3(32, C / 16), 0, stream>>>(feat, W2, hlin);
        long nc = (long)n * C;
        self_init_kernel<<<(int)((nc + TB - 1) / TB), TB, 0, stream>>>(hlin, isq, agg, nc, 7);
        long tot = (long)E << 5;
        scatter_edges_kernel<<<(int)((tot + TB - 1) / TB), TB, 0, stream>>>(
            hlin, src, dst, isq, agg, tot, C, 5);
        bias_act_kernel<<<(int)((nc + TB - 1) / TB), TB, 0, stream>>>(agg, b2, feat, nc, C - 1, 1);
    }

    // --- layer 3: feat -> d_out (no relu) ---
    {
        const int C = COUT;                 // 64
        gemm_wmma_bf16_kernel<128, 64><<<row_tiles, dim3(32, C / 16), 0, stream>>>(feat, W3, hlin);
        long nc = (long)n * C;
        self_init_kernel<<<(int)((nc + TB - 1) / TB), TB, 0, stream>>>(hlin, isq, agg, nc, 6);
        long tot = (long)E << 4;            // C/4 = 16 lanes per edge
        scatter_edges_kernel<<<(int)((tot + TB - 1) / TB), TB, 0, stream>>>(
            hlin, src, dst, isq, agg, tot, C, 4);
        bias_act_kernel<<<(int)((nc + TB - 1) / TB), TB, 0, stream>>>(
            agg, b3, (float*)d_out, nc, C - 1, 0);
    }
}
